// MoELayer_31559419691511
// MI455X (gfx1250) — matrix operand order repaired and checked by reading, not compiled, not run on hardware
//
#include <hip/hip_runtime.h>
#include <hip/hip_bf16.h>

typedef __attribute__((ext_vector_type(16))) __bf16 v16bf;
typedef __attribute__((ext_vector_type(8)))  __bf16 v8bf;
typedef __attribute__((ext_vector_type(8)))  float  v8f;

#define N_TOK  8192
#define DMODEL 2048
#define DINTER 2048
#define NEXP   8
#define KDIM   2048   // inner dim of every GEMM in this problem
#define NCOL   2048   // output cols of every GEMM in this problem

#define BM 128
#define BN 64
#define BK 32
#define LDT 40   // padded LDS row stride (bf16 elems): 80B, 16B aligned, conflict-free

__device__ __forceinline__ void splitf(float x, __bf16& h, __bf16& l) {
  h = (__bf16)x;                 // RNE to bf16
  l = (__bf16)(x - (float)h);    // residual
}

__device__ __forceinline__ float silu_fast(float v) {
  // v / (1+exp(-v)) with hardware v_rcp_f32 (TRANS, co-issues with WMMA)
  return v * __builtin_amdgcn_rcpf(1.f + __expf(-v));
}

// Build a v16bf fragment from two 16-byte LDS vectors.
__device__ __forceinline__ v16bf load_frag(const __bf16* base, int k0, int k1) {
  v8bf a = *(const v8bf*)(base + k0);
  v8bf b = *(const v8bf*)(base + k1);
  v16bf f;
#pragma unroll
  for (int t = 0; t < 8; ++t) { f[t] = a[t]; f[t + 8] = b[t]; }
  return f;
}

// C[M,2048] = act(A[M,2048] @ W[2048,2048]^T), fp32 in/out, BF16x3 WMMA inside.
template <bool ACT>
__global__ __launch_bounds__(256) void gemm_bf16x3(const float* __restrict__ A,
                                                   const float* __restrict__ W,
                                                   float* __restrict__ C) {
  __shared__ __align__(16) __bf16 Ah[BM][LDT];
  __shared__ __align__(16) __bf16 Al[BM][LDT];
  __shared__ __align__(16) __bf16 Bh[BN][LDT];
  __shared__ __align__(16) __bf16 Bl[BN][LDT];

  const int tid  = threadIdx.x;
  const int lane = tid & 31;
  const int wv   = tid >> 5;      // 8 waves
  const int wm   = wv & 3;        // 4 row groups of 32 rows
  const int wn   = wv >> 2;       // 2 col groups of 32 cols
  const int blockM = blockIdx.y * BM;
  const int blockN = blockIdx.x * BN;

  // per-thread staging coordinates (fixed across K steps)
  int arow[4], acol[4], brow[2], bcol[2];
#pragma unroll
  for (int it = 0; it < 4; ++it) {
    int id = tid + 256 * it;
    arow[it] = id >> 3; acol[it] = (id & 7) << 2;
  }
#pragma unroll
  for (int it = 0; it < 2; ++it) {
    int id = tid + 256 * it;
    brow[it] = id >> 3; bcol[it] = (id & 7) << 2;
  }

  v8f acc[2][2];
#pragma unroll
  for (int i = 0; i < 2; ++i)
#pragma unroll
    for (int j = 0; j < 2; ++j)
#pragma unroll
      for (int c = 0; c < 8; ++c) acc[i][j][c] = 0.f;

  const int r   = lane & 15;
  const int akb = (lane < 16) ? 0 : 8;    // A: low lanes K0..7/16..23, high lanes K8..15/24..31
  const int bkb = (lane < 16) ? 0 : 8;    // B: low lanes K0..7/16..23, high lanes K8..15/24..31

  // ---- software pipeline: prefetch tile 0 ----
  float4 ga[4], gb[2];
#pragma unroll
  for (int it = 0; it < 4; ++it)
    ga[it] = *(const float4*)(A + (size_t)(blockM + arow[it]) * KDIM + acol[it]);
#pragma unroll
  for (int it = 0; it < 2; ++it)
    gb[it] = *(const float4*)(W + (size_t)(blockN + brow[it]) * KDIM + bcol[it]);

  for (int k0 = 0; k0 < KDIM; k0 += BK) {
    __syncthreads();   // previous iteration's LDS reads done
    // split fp32 into hi/lo bf16 planes once per block tile
#pragma unroll
    for (int it = 0; it < 4; ++it) {
      const float* g = (const float*)&ga[it];
#pragma unroll
      for (int j = 0; j < 4; ++j) splitf(g[j], Ah[arow[it]][acol[it] + j], Al[arow[it]][acol[it] + j]);
    }
#pragma unroll
    for (int it = 0; it < 2; ++it) {
      const float* g = (const float*)&gb[it];
#pragma unroll
      for (int j = 0; j < 4; ++j) splitf(g[j], Bh[brow[it]][bcol[it] + j], Bl[brow[it]][bcol[it] + j]);
    }
    __syncthreads();

    // prefetch next tile's global data while WMMAs run
    const int kn = k0 + BK;
    if (kn < KDIM) {
#pragma unroll
      for (int it = 0; it < 4; ++it)
        ga[it] = *(const float4*)(A + (size_t)(blockM + arow[it]) * KDIM + kn + acol[it]);
#pragma unroll
      for (int it = 0; it < 2; ++it)
        gb[it] = *(const float4*)(W + (size_t)(blockN + brow[it]) * KDIM + kn + bcol[it]);
    }

    v16bf ah[2], al2[2], bh[2], bl2[2];
#pragma unroll
    for (int mi = 0; mi < 2; ++mi) {
      int rowm = wm * 32 + mi * 16 + r;
      ah[mi]  = load_frag(&Ah[rowm][0], akb, akb + 16);
      al2[mi] = load_frag(&Al[rowm][0], akb, akb + 16);
    }
#pragma unroll
    for (int ni = 0; ni < 2; ++ni) {
      int rown = wn * 32 + ni * 16 + r;
      bh[ni]  = load_frag(&Bh[rown][0], bkb, bkb + 16);
      bl2[ni] = load_frag(&Bl[rown][0], bkb, bkb + 16);
    }
    // BF16x3: hi*hi + hi*lo + lo*hi, fp32 accumulate
#pragma unroll
    for (int mi = 0; mi < 2; ++mi)
#pragma unroll
      for (int ni = 0; ni < 2; ++ni) {
        acc[mi][ni] = __builtin_amdgcn_wmma_f32_16x16x32_bf16(
            false, ah[mi], false, bh[ni], (short)0, acc[mi][ni], false, false);
        acc[mi][ni] = __builtin_amdgcn_wmma_f32_16x16x32_bf16(
            false, ah[mi], false, bl2[ni], (short)0, acc[mi][ni], false, false);
        acc[mi][ni] = __builtin_amdgcn_wmma_f32_16x16x32_bf16(
            false, al2[mi], false, bh[ni], (short)0, acc[mi][ni], false, false);
      }
  }

  const int halfM = (lane >> 4) << 3;   // C/D layout: high lanes hold M+8
#pragma unroll
  for (int mi = 0; mi < 2; ++mi)
#pragma unroll
    for (int ni = 0; ni < 2; ++ni) {
      int col  = blockN + wn * 32 + ni * 16 + r;
      int rowb = blockM + wm * 32 + mi * 16 + halfM;
#pragma unroll
      for (int rr = 0; rr < 8; ++rr) {
        float v = acc[mi][ni][rr];
        if (ACT) v = silu_fast(v);
        C[(size_t)(rowb + rr) * NCOL + col] = v;
      }
    }
}

// One wave32 per token: logits -> softmax -> top2 -> per-expert gate row Wg[N,8]
__global__ __launch_bounds__(256) void router_kernel(const float* __restrict__ x,
                                                     const float* __restrict__ wr,
                                                     float* __restrict__ Wg) {
  const int lane = threadIdx.x & 31;
  const int wv   = threadIdx.x >> 5;
  const int tok  = blockIdx.x * 8 + wv;
  float acc[NEXP];
#pragma unroll
  for (int e = 0; e < NEXP; ++e) acc[e] = 0.f;
  const float* xr = x + (size_t)tok * DMODEL;
  for (int d = lane; d < DMODEL; d += 32) {
    float xv = xr[d];
#pragma unroll
    for (int e = 0; e < NEXP; ++e) acc[e] = fmaf(xv, wr[e * DMODEL + d], acc[e]);
  }
#pragma unroll
  for (int e = 0; e < NEXP; ++e)
    for (int off = 16; off; off >>= 1) acc[e] += __shfl_xor(acc[e], off, 32);
  if (lane == 0) {
    float m = acc[0];
#pragma unroll
    for (int e = 1; e < NEXP; ++e) m = fmaxf(m, acc[e]);
    float ex[NEXP];
#pragma unroll
    for (int e = 0; e < NEXP; ++e) ex[e] = __expf(acc[e] - m);
    int i1 = 0; float v1 = ex[0];
#pragma unroll
    for (int e = 1; e < NEXP; ++e) if (ex[e] > v1) { v1 = ex[e]; i1 = e; }
    int i2 = -1; float v2 = -1.f;
#pragma unroll
    for (int e = 0; e < NEXP; ++e) if (e != i1 && ex[e] > v2) { v2 = ex[e]; i2 = e; }
    float inv = __builtin_amdgcn_rcpf(v1 + v2);  // softmax denom cancels in gate norm
    float wg[NEXP];
#pragma unroll
    for (int e = 0; e < NEXP; ++e) wg[e] = 0.f;
    wg[i1] += v1 * inv;
    wg[i2] += v2 * inv;
#pragma unroll
    for (int e = 0; e < NEXP; ++e) Wg[(size_t)tok * NEXP + e] = wg[e];
  }
}

__global__ void zero_kernel(float4* __restrict__ p, size_t n4) {
  size_t i = (size_t)blockIdx.x * blockDim.x + threadIdx.x;
  size_t stride = (size_t)gridDim.x * blockDim.x;
  float4 z; z.x = z.y = z.z = z.w = 0.f;
  for (; i < n4; i += stride) p[i] = z;
}

// out rows 0..7 = Wg^T @ h  ([8 x N] @ [N x D])
__global__ __launch_bounds__(256) void reduce_kernel(const float* __restrict__ h,
                                                     const float* __restrict__ Wg,
                                                     float* __restrict__ out) {
  __shared__ float wgs[256][NEXP];
  const int col = blockIdx.x * 256 + threadIdx.x;
  float acc[NEXP] = {};
  for (int i0 = 0; i0 < N_TOK; i0 += 256) {
#pragma unroll
    for (int e = 0; e < NEXP; ++e)
      wgs[threadIdx.x][e] = Wg[(size_t)(i0 + threadIdx.x) * NEXP + e];
    __syncthreads();
    for (int ii = 0; ii < 256; ++ii) {
      float hv = h[(size_t)(i0 + ii) * DMODEL + col];
#pragma unroll
      for (int e = 0; e < NEXP; ++e) acc[e] = fmaf(wgs[ii][e], hv, acc[e]);
    }
    __syncthreads();
  }
#pragma unroll
  for (int e = 0; e < NEXP; ++e) out[(size_t)e * DMODEL + col] = acc[e];
}

extern "C" void kernel_launch(void* const* d_in, const int* in_sizes, int n_in,
                              void* d_out, int out_size, void* d_ws, size_t ws_size,
                              hipStream_t stream) {
  const float* x  = (const float*)d_in[0];
  const float* w1 = (const float*)d_in[1];
  const float* w3 = (const float*)d_in[2];
  const float* w2 = (const float*)d_in[3];
  const float* wr = (const float*)d_in[4];
  float* out = (float*)d_out;

  // ws layout: bufA (h1 then h3, 64MB) | bufB (h2, 64MB) | Wg (256KB)
  float* bufA = (float*)d_ws;
  float* bufB = bufA + (size_t)N_TOK * DINTER;
  float* Wg   = bufB + (size_t)N_TOK * DINTER;

  router_kernel<<<N_TOK / 8, 256, 0, stream>>>(x, wr, Wg);
  zero_kernel<<<2048, 256, 0, stream>>>((float4*)out, (size_t)out_size / 4);

  dim3 blk(256);
  dim3 g(NCOL / BN, N_TOK / BM);
  gemm_bf16x3<true ><<<g, blk, 0, stream>>>(x,    w1, bufA);
  gemm_bf16x3<true ><<<g, blk, 0, stream>>>(bufA, w3, bufB);
  gemm_bf16x3<false><<<g, blk, 0, stream>>>(bufB, w2, bufA);

  reduce_kernel<<<DMODEL / 256, 256, 0, stream>>>(bufA, Wg, out);
}
